// DGT_Rank1_75634374083399
// MI455X (gfx1250) — matrix operand order repaired and checked by reading, not compiled, not run on hardware
//
#include <hip/hip_runtime.h>
#include <cstddef>
#include <cstdint>

// ---------------- model dims ----------------
#define BSZ   4
#define HLEN  25
#define FLEN  275
#define DIM   64
#define NHEAD 4
#define HDIM  16
#define DFF   2048

typedef __attribute__((ext_vector_type(16))) __bf16        v16bf;
typedef __attribute__((ext_vector_type(8)))  float         v8f;
typedef __attribute__((ext_vector_type(4)))  unsigned int  v4ui;

__device__ __forceinline__ unsigned short f2bf(float f) {
  // round-to-nearest-even f32 -> bf16 (bits)
  unsigned int u = __float_as_uint(f);
  unsigned int r = u + 0x7FFFu + ((u >> 16) & 1u);
  return (unsigned short)(r >> 16);
}

// ---------------------------------------------------------------------------
// Weight convert + transpose: W (K x N, f32) -> WT (Npad x K, bf16 bits).
// Rows n >= N are left as-is (reads of them only feed unstored WMMA columns).
// ---------------------------------------------------------------------------
__global__ void k_cvtT(const float* __restrict__ W, unsigned short* __restrict__ WT,
                       int K, int N) {
  size_t i = (size_t)blockIdx.x * blockDim.x + threadIdx.x;
  if (i >= (size_t)K * N) return;
  int k = (int)(i / N), n = (int)(i % N);
  WT[(size_t)n * K + k] = f2bf(W[i]);
}

// ---------------------------------------------------------------------------
// WMMA GEMM, all-bf16 operands, f32 accumulate:
//   C[M,N] (f32, optional) / Cbf[M,N] (bf16, optional) = act(A @ W + bias)
//   A  : bf16 row-major (M_pad x K), K % 64 == 0, rows padded to 16
//   WT : bf16 (Npad x K) transposed weights, Npad % 64 == 0
// A tiles (16x64) are staged to LDS with the CDNA5 async copy engine
// (global_load_async_to_lds_b128, ASYNCcnt), double-buffered so the next
// tile's DMA overlaps the current tile's two v_wmma_f32_16x16x32_bf16.
// 4 waves/block; wave w owns the 16x16 tile at n0 = blockIdx.y*64 + w*16.
// ---------------------------------------------------------------------------
__global__ void k_gemm(const unsigned short* __restrict__ Abf,
                       const unsigned short* __restrict__ WT,
                       const float* __restrict__ bias,
                       float* __restrict__ C, unsigned short* __restrict__ Cbf,
                       int M, int K, int N, int relu) {
  __shared__ alignas(16) unsigned short As[2][16][72];   // 16x64 bf16, 144B row stride
  const int tid  = threadIdx.x;
  const int wave = tid >> 5;
  const int lane = tid & 31;
  const int m0 = blockIdx.x * 16;
  const int n0 = blockIdx.y * 64 + wave * 16;
  const int mrow = lane & 15;
  const int col  = n0 + (lane & 15);
  const int ko  = (lane < 16) ? 0 : 8;    // A K-chunk interleave (ISA 7.12.2)
  const int ko2 = (lane < 16) ? 0 : 16;   // B K-half

  // per-thread async-copy lane: row = tid/8, 16B chunk = (tid%8)*8 elems
  const int srow = tid >> 3;
  const int schunk = (tid & 7) * 8;
  unsigned int lds0 = (unsigned int)(uintptr_t)&As[0][srow][schunk];
  unsigned int lds1 = (unsigned int)(uintptr_t)&As[1][srow][schunk];
  const unsigned short* agp = Abf + (size_t)(m0 + srow) * K + schunk;

  const int KT = K >> 6;
  v8f acc = {};

  // issue tile 0
  asm volatile("global_load_async_to_lds_b128 %0, %1, off"
               :: "v"(lds0), "v"(agp) : "memory");

  for (int kt = 0; kt < KT; ++kt) {
    if (kt + 1 < KT) {
      const unsigned short* p = agp + (size_t)(kt + 1) * 64;
      unsigned int dst = ((kt + 1) & 1) ? lds1 : lds0;
      asm volatile("global_load_async_to_lds_b128 %0, %1, off"
                   :: "v"(dst), "v"(p) : "memory");
      __builtin_prefetch(WT + (size_t)col * K + (size_t)(kt + 1) * 64, 0, 2);
      asm volatile("s_wait_asynccnt 0x1" ::: "memory");   // tile kt landed
    } else {
      asm volatile("s_wait_asynccnt 0x0" ::: "memory");
    }
    __syncthreads();

    const v4ui* arow = (const v4ui*)(&As[kt & 1][mrow][0]);
#pragma unroll
    for (int s = 0; s < 2; ++s) {
      const int ks = s * 32;
      union { v16bf v; v4ui q[2]; } a, b;
      a.q[0] = arow[(ks + ko) >> 3];        // 16B ds_load
      a.q[1] = arow[(ks + ko + 16) >> 3];
      const v4ui* bp = (const v4ui*)(WT + (size_t)col * K + (size_t)kt * 64 + ks + ko);
      b.q[0] = bp[0];                       // 16B global_load
      b.q[1] = bp[2];
      acc = __builtin_amdgcn_wmma_f32_16x16x32_bf16(
          false, a.v, false, b.v, (short)0, acc, false, false);
    }
    __syncthreads();
  }

  const int rbase = m0 + ((lane >> 4) << 3);
#pragma unroll
  for (int r = 0; r < 8; ++r) {
    int row = rbase + r;
    if (row < M && col < N) {
      float v = acc[r];
      if (bias) v += bias[col];
      if (relu) v = fmaxf(v, 0.0f);
      if (C)   C[(size_t)row * N + col] = v;
      if (Cbf) Cbf[(size_t)row * N + col] = f2bf(v);
    }
  }
}

// ---------------------------------------------------------------------------
// Streaming-softmax attention. q,k,v f32 (Bq*L, 64), head h in cols
// [16h,16h+16). One thread per (b,h,i) row. Output written as bf16 (feeds
// the O-projection GEMM directly). tmask >= 0 -> keys j<=tmask only.
// ---------------------------------------------------------------------------
__global__ void k_attn(const float* __restrict__ q, const float* __restrict__ k,
                       const float* __restrict__ v, unsigned short* __restrict__ o,
                       int Lq, int Lk, int tmask) {
  int idx = blockIdx.x * blockDim.x + threadIdx.x;
  int total = BSZ * NHEAD * Lq;
  if (idx >= total) return;
  int i = idx % Lq;
  int h = (idx / Lq) % NHEAD;
  int b = idx / (Lq * NHEAD);
  const float* qr = q + ((size_t)(b * Lq + i)) * DIM + h * HDIM;
  float qv[HDIM];
#pragma unroll
  for (int d = 0; d < HDIM; ++d) qv[d] = qr[d] * 0.25f;   // 1/sqrt(16)
  int kmax = (tmask >= 0) ? ((tmask + 1 < Lk) ? tmask + 1 : Lk) : Lk;
  float mx = -3.0e38f, sum = 0.0f;
  float ov[HDIM];
#pragma unroll
  for (int d = 0; d < HDIM; ++d) ov[d] = 0.0f;
  for (int j = 0; j < kmax; ++j) {
    const float* kr = k + ((size_t)(b * Lk + j)) * DIM + h * HDIM;
    float s = 0.0f;
#pragma unroll
    for (int d = 0; d < HDIM; ++d) s += qv[d] * kr[d];
    float nm = fmaxf(mx, s);
    float corr = __expf(mx - nm);
    float w = __expf(s - nm);
    sum = sum * corr + w;
    const float* vr = v + ((size_t)(b * Lk + j)) * DIM + h * HDIM;
#pragma unroll
    for (int d = 0; d < HDIM; ++d) ov[d] = ov[d] * corr + w * vr[d];
    mx = nm;
  }
  float inv = 1.0f / sum;
  unsigned short* orow = o + ((size_t)(b * Lq + i)) * DIM + h * HDIM;
#pragma unroll
  for (int d = 0; d < HDIM; ++d) orow[d] = f2bf(ov[d] * inv);
}

// ---------------------------------------------------------------------------
// out(f32) [+ outbf(bf16)] = LayerNorm(x (+ r)) * g + b, last dim 64.
// One wave32 per row.
// ---------------------------------------------------------------------------
__global__ void k_add_ln(const float* __restrict__ x, const float* __restrict__ r,
                         const float* __restrict__ g, const float* __restrict__ bta,
                         float* __restrict__ out, unsigned short* __restrict__ outbf,
                         int M) {
  int wave = threadIdx.x >> 5, lane = threadIdx.x & 31;
  int row = blockIdx.x * 4 + wave;
  if (row >= M) return;
  const float* xr = x + (size_t)row * DIM;
  float v0 = xr[lane], v1 = xr[lane + 32];
  if (r) {
    const float* rr = r + (size_t)row * DIM;
    v0 += rr[lane];
    v1 += rr[lane + 32];
  }
  float s = v0 + v1, s2 = v0 * v0 + v1 * v1;
#pragma unroll
  for (int off = 16; off > 0; off >>= 1) {
    s  += __shfl_xor(s, off, 32);
    s2 += __shfl_xor(s2, off, 32);
  }
  float mu = s * (1.0f / DIM);
  float var = s2 * (1.0f / DIM) - mu * mu;
  float rs = rsqrtf(var + 1e-5f);
  float o0 = (v0 - mu) * rs * g[lane]      + bta[lane];
  float o1 = (v1 - mu) * rs * g[lane + 32] + bta[lane + 32];
  out[(size_t)row * DIM + lane]      = o0;
  out[(size_t)row * DIM + lane + 32] = o1;
  if (outbf) {
    outbf[(size_t)row * DIM + lane]      = f2bf(o0);
    outbf[(size_t)row * DIM + lane + 32] = f2bf(o1);
  }
}

// embed: x[b,t,:] = xh[b,t]*proj_w + proj_b + pos[t,:]  (f32 + bf16 copies)
__global__ void k_embed(const float* __restrict__ xh, const float* __restrict__ pw,
                        const float* __restrict__ pb, const float* __restrict__ pos,
                        float* __restrict__ x, unsigned short* __restrict__ xbf) {
  int i = blockIdx.x * blockDim.x + threadIdx.x;
  if (i >= BSZ * HLEN * DIM) return;
  int d = i & 63;
  int t = (i >> 6) % HLEN;
  int b = i / (DIM * HLEN);
  float v = xh[b * HLEN + t] * pw[d] + pb[d] + pos[t * DIM + d];
  x[i] = v;
  xbf[i] = f2bf(v);
}

// ---------------------------------------------------------------------------
// Fuzzy c-means (m=2), whole solve in one block. data (4,1600), u0 (3,4).
// cmap[1600] = mean over clusters of final centers.
// ---------------------------------------------------------------------------
__global__ void k_fcm(const float* __restrict__ data, const float* __restrict__ u0,
                      float* __restrict__ cmap) {
  __shared__ float sd[BSZ * HLEN * DIM];   // 6400
  __shared__ float cn[3 * HLEN * DIM];     // 4800
  __shared__ float su[12], sun[12], sd2[12], red[256];
  __shared__ float sdiff;
  const int tid = threadIdx.x;
  for (int i = tid; i < 6400; i += 256) sd[i] = data[i];
  if (tid < 12) su[tid] = u0[tid];
  __syncthreads();
  for (int iter = 0; iter < 1000; ++iter) {
    float um[12], den[3];
#pragma unroll
    for (int c = 0; c < 3; ++c) {
      den[c] = 0.0f;
#pragma unroll
      for (int b = 0; b < 4; ++b) {
        um[c * 4 + b] = su[c * 4 + b] * su[c * 4 + b];
        den[c] += um[c * 4 + b];
      }
    }
    for (int j = tid; j < 1600; j += 256) {
#pragma unroll
      for (int c = 0; c < 3; ++c) {
        float acc = 0.0f;
#pragma unroll
        for (int b = 0; b < 4; ++b) acc += um[c * 4 + b] * sd[b * 1600 + j];
        cn[c * 1600 + j] = acc / den[c];
      }
    }
    __syncthreads();
    for (int cb = 0; cb < 12; ++cb) {
      int c = cb >> 2, b = cb & 3;
      float p = 0.0f;
      for (int j = tid; j < 1600; j += 256) {
        float df = sd[b * 1600 + j] - cn[c * 1600 + j];
        p += df * df;
      }
      red[tid] = p;
      __syncthreads();
      for (int off = 128; off > 0; off >>= 1) {
        if (tid < off) red[tid] += red[tid + off];
        __syncthreads();
      }
      if (tid == 0) sd2[cb] = red[0];
      __syncthreads();
    }
    if (tid == 0) {
      float un[12];
#pragma unroll
      for (int cb = 0; cb < 12; ++cb) {
        float dd = fmaxf(sqrtf(sd2[cb]), 1e-12f);
        un[cb] = 1.0f / (dd * dd);          // d^(-2/(m-1)), m=2
      }
#pragma unroll
      for (int b = 0; b < 4; ++b) {
        float cs = un[b] + un[4 + b] + un[8 + b];
#pragma unroll
        for (int c = 0; c < 3; ++c) un[c * 4 + b] /= cs;
      }
      float df2 = 0.0f;
#pragma unroll
      for (int cb = 0; cb < 12; ++cb) {
        float d = un[cb] - su[cb];
        df2 += d * d;
        sun[cb] = un[cb];
      }
      sdiff = sqrtf(df2);
    }
    __syncthreads();
    if (tid < 12) su[tid] = sun[tid];
    __syncthreads();
    if (sdiff <= 1e-5f) break;
  }
  float um[12], den[3];
#pragma unroll
  for (int c = 0; c < 3; ++c) {
    den[c] = 0.0f;
#pragma unroll
    for (int b = 0; b < 4; ++b) {
      um[c * 4 + b] = su[c * 4 + b] * su[c * 4 + b];
      den[c] += um[c * 4 + b];
    }
  }
  for (int j = tid; j < 1600; j += 256) {
    float mean = 0.0f;
#pragma unroll
    for (int c = 0; c < 3; ++c) {
      float acc = 0.0f;
#pragma unroll
      for (int b = 0; b < 4; ++b) acc += um[c * 4 + b] * sd[b * 1600 + j];
      mean += acc / den[c];
    }
    cmap[j] = mean * (1.0f / 3.0f);
  }
}

// fused[b,t,:] = hid[b,t,:] + mot[b,t] + cmap[t,:]  (f32 + bf16)
__global__ void k_fuse(const float* __restrict__ hid, const float* __restrict__ mot,
                       const float* __restrict__ cmap, float* __restrict__ fused,
                       unsigned short* __restrict__ fusedbf) {
  int i = blockIdx.x * blockDim.x + threadIdx.x;
  if (i >= BSZ * HLEN * DIM) return;
  int d = i & 63;
  int t = (i >> 6) % HLEN;
  int b = i / (DIM * HLEN);
  float v = hid[i] + mot[b * HLEN + t] + cmap[t * DIM + d];
  fused[i] = v;
  fusedbf[i] = f2bf(v);
}

__global__ void k_zero_f32(float* __restrict__ p, int n) {
  int i = blockIdx.x * blockDim.x + threadIdx.x;
  if (i < n) p[i] = 0.0f;
}
__global__ void k_zero_u16(unsigned short* __restrict__ p, int n) {
  int i = blockIdx.x * blockDim.x + threadIdx.x;
  if (i < n) p[i] = 0;
}

// outs_bf[:,t,:] = x[:,t,:];  buf[f32+bf16][:,min(t+1,274),:] = x[:,t,:]
__global__ void k_slice(const float* __restrict__ x, float* __restrict__ buf,
                        unsigned short* __restrict__ bufbf,
                        unsigned short* __restrict__ outsbf, int t) {
  int i = threadIdx.x;                       // 256 == BSZ*DIM
  int b = i >> 6, d = i & 63;
  size_t src = ((size_t)b * FLEN + t) * DIM + d;
  float val = x[src];
  outsbf[src] = f2bf(val);
  int t2 = (t + 1 < FLEN) ? t + 1 : FLEN - 1;
  size_t dst = ((size_t)b * FLEN + t2) * DIM + d;
  buf[dst] = val;
  bufbf[dst] = f2bf(val);
}

// ---------------------------------------------------------------------------
// Host orchestration
// ---------------------------------------------------------------------------
struct MhaDev {
  const unsigned short *wqT, *wkT, *wvT, *woT;
  const float *bq, *bk, *bv, *bo;
};
struct EncDev {
  MhaDev sa;
  const unsigned short *w1T, *w2T;
  const float *b1, *b2, *ln1g, *ln1b, *ln2g, *ln2b;
};
struct DecDev {
  MhaDev sa, ca;
  const unsigned short *w1T, *w2T;
  const float *b1, *b2, *ln1g, *ln1b, *ln2g, *ln2b, *ln3g, *ln3b;
};

extern "C" void kernel_launch(void* const* d_in, const int* in_sizes, int n_in,
                              void* d_out, int out_size, void* d_ws, size_t ws_size,
                              hipStream_t stream) {
  (void)in_sizes; (void)n_in; (void)out_size; (void)ws_size;
  auto F = [&](int i) -> const float* { return (const float*)d_in[i]; };

  // ---- workspace bump allocator ----
  char* wsp = (char*)d_ws;
  size_t woff = 0;
  auto alloc = [&](size_t bytes) -> void* {
    size_t a = (woff + 255) & ~(size_t)255;
    woff = a + bytes;
    return (void*)(wsp + a);
  };
  auto allocF = [&](size_t n) -> float* { return (float*)alloc(n * sizeof(float)); };
  auto allocH = [&](size_t n) -> unsigned short* {
    return (unsigned short*)alloc(n * sizeof(unsigned short));
  };

  // ---- weight conversion (f32 KxN -> bf16 Npad x K, Npad % 64 == 0) ----
  auto cvt = [&](const float* W, int K, int N) -> unsigned short* {
    int Np = (N + 63) & ~63;
    unsigned short* dst = allocH((size_t)Np * K);
    size_t total = (size_t)K * N;
    k_cvtT<<<dim3((unsigned)((total + 255) / 256)), 256, 0, stream>>>(W, dst, K, N);
    return dst;
  };

  // ---- input index maps (jax tree flatten: dict keys sorted, lists in order) --
  auto mk_mha = [&](int base) -> MhaDev {   // bk,bo,bq,bv,wk,wo,wq,wv
    MhaDev m;
    m.bk = F(base + 0); m.bo = F(base + 1); m.bq = F(base + 2); m.bv = F(base + 3);
    m.wkT = cvt(F(base + 4), DIM, DIM);
    m.woT = cvt(F(base + 5), DIM, DIM);
    m.wqT = cvt(F(base + 6), DIM, DIM);
    m.wvT = cvt(F(base + 7), DIM, DIM);
    return m;
  };
  auto mk_enc = [&](int base) -> EncDev {   // b1,b2,ln1b,ln1g,ln2b,ln2g,sa(8),w1,w2
    EncDev e;
    e.b1 = F(base + 0);  e.b2 = F(base + 1);
    e.ln1b = F(base + 2); e.ln1g = F(base + 3);
    e.ln2b = F(base + 4); e.ln2g = F(base + 5);
    e.sa = mk_mha(base + 6);
    e.w1T = cvt(F(base + 14), DIM, DFF);
    e.w2T = cvt(F(base + 15), DFF, DIM);
    return e;
  };
  auto mk_dec = [&](int base) -> DecDev {   // b1,b2,ca(8),ln*(6),sa(8),w1,w2
    DecDev d;
    d.b1 = F(base + 0); d.b2 = F(base + 1);
    d.ca = mk_mha(base + 2);
    d.ln1b = F(base + 10); d.ln1g = F(base + 11);
    d.ln2b = F(base + 12); d.ln2g = F(base + 13);
    d.ln3b = F(base + 14); d.ln3g = F(base + 15);
    d.sa = mk_mha(base + 16);
    d.w1T = cvt(F(base + 24), DIM, DFF);
    d.w2T = cvt(F(base + 25), DFF, DIM);
    return d;
  };

  const float* xh     = F(0);
  const float* mot    = F(1);
  const float* proj_w = F(2);
  const float* proj_b = F(3);
  const float* pos    = F(4);
  DecDev s2d0 = mk_dec(5),   s2d1 = mk_dec(31);
  const float* s2_dec_nb = F(57); const float* s2_dec_ng = F(58);
  EncDev s2e0 = mk_enc(59),  s2e1 = mk_enc(75);
  const float* s2_enc_nb = F(91); const float* s2_enc_ng = F(92);
  DecDev agd0 = mk_dec(93),  agd1 = mk_dec(119);
  EncDev age0 = mk_enc(145), age1 = mk_enc(161);
  const float* reg_b1 = F(177);
  const float* reg_b2 = F(178);
  unsigned short* regw1T = cvt(F(179), FLEN * DIM, 128);  // (17600,128)
  unsigned short* regw2T = cvt(F(180), 128, FLEN);
  const float* u0 = F(181);

  // ---- activation buffers (row counts padded to 16 for WMMA A reads) ----
  const int MF = BSZ * FLEN;            // 1100
  const int MH = BSZ * HLEN;            // 100
  const int MFP = 1104;                 // padded rows
  float* qb    = allocF((size_t)MF * DIM);
  float* kb    = allocF((size_t)MF * DIM);
  float* vb    = allocF((size_t)MF * DIM);
  float* ob    = allocF((size_t)MF * DIM);
  float* x1b   = allocF((size_t)MF * DIM);
  float* x2b   = allocF((size_t)MF * DIM);
  float* xemb  = allocF((size_t)MH * DIM);
  float* memb  = allocF((size_t)MH * DIM);
  float* hid   = allocF((size_t)MH * DIM);
  float* cmap  = allocF((size_t)HLEN * DIM);
  float* fused = allocF((size_t)MH * DIM);
  float* buf   = allocF((size_t)MF * DIM);
  float* xA    = allocF((size_t)MF * DIM);
  float* kc[4]; float* vc[4];
  for (int i = 0; i < 4; ++i) { kc[i] = allocF((size_t)MH * DIM); vc[i] = allocF((size_t)MH * DIM); }
  // bf16 companions (GEMM A operands)
  unsigned short* xembbf  = allocH((size_t)MFP * DIM);
  unsigned short* membbf  = allocH((size_t)MFP * DIM);
  unsigned short* fusedbf = allocH((size_t)MFP * DIM);
  unsigned short* x1bf    = allocH((size_t)MFP * DIM);
  unsigned short* x2bf    = allocH((size_t)MFP * DIM);
  unsigned short* abf     = allocH((size_t)MFP * DIM);
  unsigned short* bufbf   = allocH((size_t)MFP * DIM);
  unsigned short* xAbf    = allocH((size_t)MFP * DIM);
  unsigned short* hbf     = allocH((size_t)MFP * DFF);
  unsigned short* outsbf  = allocH((size_t)16 * FLEN * DIM);   // (4,17600) pad->16
  unsigned short* r1bf    = allocH((size_t)16 * 128);

  // ---- launch helpers ----
  auto gemm = [&](const unsigned short* A, const unsigned short* WT, const float* bias,
                  float* C, unsigned short* Cbf, int M, int K, int N, int relu) {
    dim3 g((unsigned)((M + 15) / 16), (unsigned)((N + 63) / 64));
    k_gemm<<<g, 128, 0, stream>>>(A, WT, bias, C, Cbf, M, K, N, relu);
  };
  auto attn = [&](const float* q, const float* k, const float* v, unsigned short* o,
                  int Lq, int Lk, int tmask) {
    int total = BSZ * NHEAD * Lq;
    k_attn<<<(total + 127) / 128, 128, 0, stream>>>(q, k, v, o, Lq, Lk, tmask);
  };
  auto addln = [&](const float* x, const float* r, const float* g, const float* b,
                   float* out, unsigned short* outbf, int M) {
    k_add_ln<<<(M + 3) / 4, 128, 0, stream>>>(x, r, g, b, out, outbf, M);
  };
  auto mha_self = [&](const unsigned short* xbf, int L, const MhaDev& W,
                      float* out, int tmask) {
    int M = BSZ * L;
    gemm(xbf, W.wqT, W.bq, qb, nullptr, M, DIM, DIM, 0);
    gemm(xbf, W.wkT, W.bk, kb, nullptr, M, DIM, DIM, 0);
    gemm(xbf, W.wvT, W.bv, vb, nullptr, M, DIM, DIM, 0);
    attn(qb, kb, vb, abf, L, L, tmask);
    gemm(abf, W.woT, W.bo, out, nullptr, M, DIM, DIM, 0);
  };
  auto enc_layer = [&](float* x, unsigned short* xbf, int L, const EncDev& P) {
    int M = BSZ * L;
    mha_self(xbf, L, P.sa, ob, -1);
    addln(x, ob, P.ln1g, P.ln1b, x, xbf, M);
    gemm(xbf, P.w1T, P.b1, nullptr, hbf, M, DIM, DFF, 1);
    gemm(hbf, P.w2T, P.b2, ob, nullptr, M, DFF, DIM, 0);
    addln(x, ob, P.ln2g, P.ln2b, x, xbf, M);
  };
  auto dec_layer = [&](const float* xin, const unsigned short* xinbf,
                       float* xout, unsigned short* xoutbf, int L, const DecDev& P,
                       const float* kcc, const float* vcc, int Lk, int tmask) {
    int M = BSZ * L;
    mha_self(xinbf, L, P.sa, ob, tmask);
    addln(xin, ob, P.ln1g, P.ln1b, x1b, x1bf, M);
    gemm(x1bf, P.ca.wqT, P.ca.bq, qb, nullptr, M, DIM, DIM, 0);
    attn(qb, kcc, vcc, abf, L, Lk, -1);
    gemm(abf, P.ca.woT, P.ca.bo, ob, nullptr, M, DIM, DIM, 0);
    addln(x1b, ob, P.ln2g, P.ln2b, x2b, x2bf, M);
    gemm(x2bf, P.w1T, P.b1, nullptr, hbf, M, DIM, DFF, 1);
    gemm(hbf, P.w2T, P.b2, ob, nullptr, M, DFF, DIM, 0);
    addln(x2b, ob, P.ln3g, P.ln3b, xout, xoutbf, M);
  };

  // ================= pipeline =================
  // 1) embed (x and mem working copies)
  k_embed<<<(MH * DIM + 255) / 256, 256, 0, stream>>>(xh, proj_w, proj_b, pos, xemb, xembbf);
  k_embed<<<(MH * DIM + 255) / 256, 256, 0, stream>>>(xh, proj_w, proj_b, pos, memb, membbf);

  // 2) seq2 encoders -> mem
  enc_layer(memb, membbf, HLEN, s2e0);
  enc_layer(memb, membbf, HLEN, s2e1);
  addln(memb, nullptr, s2_enc_ng, s2_enc_nb, memb, membbf, MH);

  // 3) seq2 decoders (cross K/V hoisted)
  gemm(membbf, s2d0.ca.wkT, s2d0.ca.bk, kc[0], nullptr, MH, DIM, DIM, 0);
  gemm(membbf, s2d0.ca.wvT, s2d0.ca.bv, vc[0], nullptr, MH, DIM, DIM, 0);
  gemm(membbf, s2d1.ca.wkT, s2d1.ca.bk, kc[1], nullptr, MH, DIM, DIM, 0);
  gemm(membbf, s2d1.ca.wvT, s2d1.ca.bv, vc[1], nullptr, MH, DIM, DIM, 0);
  dec_layer(xemb, xembbf, xemb, xembbf, HLEN, s2d0, kc[0], vc[0], HLEN, -1);
  dec_layer(xemb, xembbf, xemb, xembbf, HLEN, s2d1, kc[1], vc[1], HLEN, -1);
  addln(xemb, nullptr, s2_dec_ng, s2_dec_nb, hid, nullptr, MH);

  // 4) fuzzy c-means -> cmap ; 5) fuse
  k_fcm<<<1, 256, 0, stream>>>(hid, u0, cmap);
  k_fuse<<<(MH * DIM + 255) / 256, 256, 0, stream>>>(hid, mot, cmap, fused, fusedbf);

  // 6) 'again' encoders: fused -> mem2 (in place), hoist cross K/V
  enc_layer(fused, fusedbf, HLEN, age0);
  enc_layer(fused, fusedbf, HLEN, age1);
  gemm(fusedbf, agd0.ca.wkT, agd0.ca.bk, kc[2], nullptr, MH, DIM, DIM, 0);
  gemm(fusedbf, agd0.ca.wvT, agd0.ca.bv, vc[2], nullptr, MH, DIM, DIM, 0);
  gemm(fusedbf, agd1.ca.wkT, agd1.ca.bk, kc[3], nullptr, MH, DIM, DIM, 0);
  gemm(fusedbf, agd1.ca.wvT, agd1.ca.bv, vc[3], nullptr, MH, DIM, DIM, 0);

  // 7) autoregressive decode: 275 sequential steps over (4,275,64)
  k_zero_f32<<<(MF * DIM + 255) / 256, 256, 0, stream>>>(buf, MF * DIM);
  k_zero_u16<<<(MFP * DIM + 255) / 256, 256, 0, stream>>>(bufbf, MFP * DIM);
  for (int t = 0; t < FLEN; ++t) {
    dec_layer(buf, bufbf, xA, xAbf, FLEN, agd0, kc[2], vc[2], HLEN, t);
    dec_layer(xA, xAbf, xA, xAbf, FLEN, agd1, kc[3], vc[3], HLEN, t);
    k_slice<<<1, 256, 0, stream>>>(xA, buf, bufbf, outsbf, t);
  }

  // 8) regression head: (4,17600)@(17600,128) relu @ (128,275)
  gemm(outsbf, regw1T, reg_b1, nullptr, r1bf, BSZ, FLEN * DIM, 128, 1);
  gemm(r1bf, regw2T, reg_b2, (float*)d_out, nullptr, BSZ, 128, FLEN, 0);
}
